// DifferentiableHMM_Centered_4964982194847
// MI455X (gfx1250) — 8-bit matrix operand order repaired and checked by reading, compiled, not run on hardware
//
#include <hip/hip_runtime.h>
#include <hip/hip_bf16.h>


typedef __attribute__((ext_vector_type(8))) int v8i;

#define NS 4096
#define NB 4096
#define NE 24576
#define NELEM (NS * NB)
#define ROWB (NB * 4)

__global__ __launch_bounds__(256)
void fused_st_gumbel_kernel(const float* __restrict__ x,
                            const float* __restrict__ gum,
                            const float* __restrict__ state_means,
                            const float* __restrict__ log_stds,
                            float* __restrict__ out,
                            unsigned char* __restrict__ oh)
{
    const int t = blockIdx.x * blockDim.x + threadIdx.x;
    if (t == 0) out[NELEM] = 0.0f;
    if (t >= NELEM) return;

    const float sm0 = state_means[0];
    const float sm2 = state_means[1];
    const float sd0 = expf(log_stds[0]) + 1e-6f;
    const float sd1 = expf(log_stds[1]) + 1e-6f;
    const float sd2 = expf(log_stds[2]) + 1e-6f;
    const float i0 = 1.0f / sd0, i1 = 1.0f / sd1, i2 = 1.0f / sd2;
    const float c0 = -logf(sd0) - 0.91893853320467274178f;
    const float c1 = -logf(sd1) - 0.91893853320467274178f;
    const float c2 = -logf(sd2) - 0.91893853320467274178f;

    const float xv = x[t];
    const float z0 = (xv - sm0) * i0;
    const float z1 = xv * i1;
    const float z2 = (xv - sm2) * i2;
    const float le0 = fmaf(-0.5f * z0, z0, c0);
    const float le1 = fmaf(-0.5f * z1, z1, c1);
    const float le2 = fmaf(-0.5f * z2, z2, c2);

    const float g0 = gum[3 * t + 0];
    const float g1 = gum[3 * t + 1];
    const float g2 = gum[3 * t + 2];
    const float y0 = (le0 + g0) * 10.0f;
    const float y1 = (le1 + g1) * 10.0f;
    const float y2 = (le2 + g2) * 10.0f;

    const float mx = fmaxf(y0, fmaxf(y1, y2));
    const float e0 = expf(y0 - mx);
    const float e1 = expf(y1 - mx);
    const float e2 = expf(y2 - mx);
    const float inv = 1.0f / (e0 + e1 + e2);
    const float p0 = e0 * inv, p1 = e1 * inv, p2 = e2 * inv;

    const int am = (p1 > p0) ? ((p2 > p1) ? 2 : 1) : ((p2 > p0) ? 2 : 0);
    const float h0 = (am == 0) ? 1.0f : 0.0f;
    const float h1 = (am == 1) ? 1.0f : 0.0f;
    const float h2 = (am == 2) ? 1.0f : 0.0f;

    const float st0 = (h0 - p0) + p0;
    const float st1 = (h1 - p1) + p1;
    const float st2 = (h2 - p2) + p2;
    out[t] = st0 * sm0 + st1 * 0.0f + st2 * sm2;

    const unsigned int pack = (am == 0 ? 1u : 0u)
                            | ((am == 1 ? 1u : 0u) << 8)
                            | ((am == 2 ? 1u : 0u) << 16);
    reinterpret_cast<unsigned int*>(oh)[t] = pack;
}

__global__ __launch_bounds__(256)
void edge_loss_wmma_kernel(const int* __restrict__ edge_index,
                           const unsigned char* __restrict__ oh,
                           float* __restrict__ loss_out)
{
    const int wave = (blockIdx.x * blockDim.x + threadIdx.x) >> 5;
    const int lane = threadIdx.x & 31;
    const int e0   = wave * 16;
    const int m    = lane & 15;
    const int half = lane >> 4;

    const size_t rowOff = (size_t)edge_index[e0 + m] * (size_t)ROWB;
    const size_t colOff = (size_t)edge_index[NE + e0 + m] * (size_t)ROWB;

    v8i acc = (v8i)(0);

    for (int kb = 0; kb < ROWB; kb += 64) {
        __builtin_prefetch(oh + rowOff + kb + 512, 0, 1);
        __builtin_prefetch(oh + colOff + kb + 512, 0, 1);

        const size_t abase = rowOff + (size_t)kb + (size_t)(half * 8);
        const uint2 a01 = *reinterpret_cast<const uint2*>(oh + abase + 0);
        const uint2 a23 = *reinterpret_cast<const uint2*>(oh + abase + 16);
        const uint2 a45 = *reinterpret_cast<const uint2*>(oh + abase + 32);
        const uint2 a67 = *reinterpret_cast<const uint2*>(oh + abase + 48);
        const v8i A = { (int)a01.x, (int)a01.y, (int)a23.x, (int)a23.y,
                        (int)a45.x, (int)a45.y, (int)a67.x, (int)a67.y };

        const size_t bbase = colOff + (size_t)kb + (size_t)(half * 8);
        const uint2 b01 = *reinterpret_cast<const uint2*>(oh + bbase + 0);
        const uint2 b23 = *reinterpret_cast<const uint2*>(oh + bbase + 16);
        const uint2 b45 = *reinterpret_cast<const uint2*>(oh + bbase + 32);
        const uint2 b67 = *reinterpret_cast<const uint2*>(oh + bbase + 48);
        const v8i B = { (int)b01.x, (int)b01.y, (int)b23.x, (int)b23.y,
                        (int)b45.x, (int)b45.y, (int)b67.x, (int)b67.y };

        acc = __builtin_amdgcn_wmma_i32_16x16x64_iu8(
             false, A,  false, B, acc,
             false,  false);
    }

    int matches = 0;
    bool isDiag = false;
    int idx = 0;
    if (lane < 8)        { isDiag = true; idx = lane; }
    else if (lane >= 24) { isDiag = true; idx = lane - 24; }
#pragma unroll
    for (int r = 0; r < 8; ++r)
        if (idx == r) matches = acc[r];

    float part = isDiag ? 2.0f * (float)(NB - matches) : 0.0f;
#pragma unroll
    for (int off = 16; off >= 1; off >>= 1)
        part += __shfl_xor(part, off, 32);

    if (lane == 0) {
        const float scale = 0.1f / ((float)NE * (float)NB * 3.0f);
        atomicAdd(loss_out, part * scale);
    }
}

extern "C" void kernel_launch(void* const* d_in, const int* in_sizes, int n_in,
                              void* d_out, int out_size, void* d_ws, size_t ws_size,
                              hipStream_t stream) {
    (void)in_sizes; (void)n_in; (void)out_size; (void)ws_size;
    const float* x   = (const float*)d_in[0];
    const int*   ei  = (const int*)d_in[1];
    const float* gum = (const float*)d_in[2];
    const float* sm  = (const float*)d_in[3];
    const float* ls  = (const float*)d_in[4];

    float* out = (float*)d_out;
    unsigned char* oh = (unsigned char*)d_ws;

    fused_st_gumbel_kernel<<<NELEM / 256, 256, 0, stream>>>(x, gum, sm, ls, out, oh);

    edge_loss_wmma_kernel<<<(NE / 16) / 8, 256, 0, stream>>>(ei, oh, out + NELEM);
}
